// QuantumEnhancedDiffusion_71854802862779
// MI455X (gfx1250) — matrix operand order repaired and checked by reading, not compiled, not run on hardware
//
#include <hip/hip_runtime.h>
#include <hip/hip_bf16.h>

typedef __attribute__((ext_vector_type(16))) __bf16 v16bf;
typedef __attribute__((ext_vector_type(8)))  __bf16 bf16x8;
typedef __attribute__((ext_vector_type(8)))  float  v8f;
typedef __attribute__((ext_vector_type(4)))  float  f32x4;

#define BATCH 32768
#define DIMD  1024
#define DIMH  512
#define NQ    8
#define NL    3
#define NDEPTH 6

// ---------------------------------------------------------------------------
// Weight prep: W (K x N, f32, row-major) -> Wt (N x Kpad, bf16, row-major)
// Kpad >= K, zero padded. This makes WMMA B-fragment loads contiguous.
// ---------------------------------------------------------------------------
__global__ void prep_wt_kernel(const float* __restrict__ W, __bf16* __restrict__ Wt,
                               int K, int N, int Kpad) {
    int idx = blockIdx.x * blockDim.x + threadIdx.x;
    if (idx >= N * Kpad) return;
    int n  = idx / Kpad;
    int kk = idx - n * Kpad;
    float v = (kk < K) ? W[(size_t)kk * N + n] : 0.0f;
    Wt[idx] = (__bf16)v;
}

// ---------------------------------------------------------------------------
// Generic WMMA bf16 GEMM:  C[M,N] = act( A[M,K] @ W[K,N] + bias[N] )
//   A: bf16 row-major (lda=K) or f32 row-major (AF32, converted inline)
//   W: pre-transposed bf16  Wt[N][K]
//   Block = 128 threads = 4 waves; wave w handles rows [64*bx + 16*w, +16),
//   columns [ (NT*16)*by , +NT*16 ).  Requires M%64==0, K%32==0, N%(NT*16)==0.
//   Per K-step: all NT B fragments are loaded into distinct registers first,
//   then the WMMA chain runs -> partial loadcnt waits, VMEM/XDL overlap.
// ---------------------------------------------------------------------------
template <int NT, bool RELU, bool AF32, bool OUTF32>
__global__ __launch_bounds__(128)
void gemm_wmma_kernel(const void* __restrict__ Av,
                      const __bf16* __restrict__ Wt,
                      const float* __restrict__ bias,
                      void* __restrict__ Cv,
                      int M, int N, int K) {
    const int lane = threadIdx.x & 31;
    const int wave = threadIdx.x >> 5;
    const int half = lane >> 4;          // 0: lanes 0-15, 1: lanes 16-31
    const int m16  = lane & 15;

    const int row0 = blockIdx.x * 64 + wave * 16;
    const int col0 = blockIdx.y * (NT * 16);

    const int r    = row0 + m16;         // A row owned by this lane
    const int khiA = half * 8;           // A-frag K sub-offset per ISA layout
    const int khiB = half * 8;           // B-frag K sub-offset per ISA layout

    v8f acc[NT];
#pragma unroll
    for (int t = 0; t < NT; ++t) acc[t] = (v8f){0.f,0.f,0.f,0.f,0.f,0.f,0.f,0.f};

    const __bf16* Ab = (const __bf16*)Av;
    const float*  Af = (const float*)Av;
    // per-lane base into Wt for this lane's column within tile 0
    const __bf16* wbase = Wt + (size_t)(col0 + m16) * K + khiB;

    for (int k = 0; k < K; k += 32) {
        // ---- B fragments: NT tiles into distinct registers ----
        v16bf bfr[NT];
#pragma unroll
        for (int t = 0; t < NT; ++t) {
            bf16x8 blo = *(const bf16x8*)(wbase + (size_t)t * 16 * K + k);
            bf16x8 bhi = *(const bf16x8*)(wbase + (size_t)t * 16 * K + k + 16);
#pragma unroll
            for (int i = 0; i < 8; ++i) { bfr[t][i] = blo[i]; bfr[t][8 + i] = bhi[i]; }
        }

        // ---- A fragment (16x32 bf16, per ISA 16-bit A layout) ----
        v16bf afrag;
        if (AF32) {
            const float* ap = Af + (size_t)r * K + k;
            if (k + 64 < K) __builtin_prefetch(ap + 64, 0, 1);
            f32x4 q0 = *(const f32x4*)(ap + khiA);
            f32x4 q1 = *(const f32x4*)(ap + khiA + 4);
            f32x4 q2 = *(const f32x4*)(ap + 16 + khiA);
            f32x4 q3 = *(const f32x4*)(ap + 16 + khiA + 4);
#pragma unroll
            for (int i = 0; i < 4; ++i) {
                afrag[i]      = (__bf16)q0[i];
                afrag[4 + i]  = (__bf16)q1[i];
                afrag[8 + i]  = (__bf16)q2[i];
                afrag[12 + i] = (__bf16)q3[i];
            }
        } else {
            const __bf16* ap = Ab + (size_t)r * K + k;
            if (k + 64 < K) __builtin_prefetch(ap + 64, 0, 1);
            bf16x8 lo = *(const bf16x8*)(ap + khiA);
            bf16x8 hi = *(const bf16x8*)(ap + 16 + khiA);
#pragma unroll
            for (int i = 0; i < 8; ++i) {
                afrag[i]     = lo[i];
                afrag[8 + i] = hi[i];
            }
        }

        // ---- WMMA chain over the column tiles ----
#pragma unroll
        for (int t = 0; t < NT; ++t) {
            acc[t] = __builtin_amdgcn_wmma_f32_16x16x32_bf16(
                false, afrag, false, bfr[t], (short)0, acc[t], false, false);
        }
    }

    // ---- epilogue: bias (+ReLU) and store per ISA C layout ----
    float* Cf  = (float*)Cv;
    __bf16* Cb = (__bf16*)Cv;
#pragma unroll
    for (int t = 0; t < NT; ++t) {
        const int col = col0 + t * 16 + m16;
        const float bb = bias ? bias[col] : 0.0f;
#pragma unroll
        for (int v = 0; v < 8; ++v) {
            const int row = row0 + v + 8 * half;
            float val = acc[t][v] + bb;
            if (RELU) val = val > 0.0f ? val : 0.0f;
            if (OUTF32) Cf[(size_t)row * N + col] = val;
            else        Cb[(size_t)row * N + col] = (__bf16)val;
        }
    }
}

// ---------------------------------------------------------------------------
// Quantum core: one thread per batch row.
//  e[b,16] -> amp softmax / phases -> state -> L x (DEPTH variational steps
//  + noise) -> probs softmax -> gumbel softmax -> meas bf16 padded to 32 cols
// ---------------------------------------------------------------------------
__global__ __launch_bounds__(256)
void quantum_kernel(const float* __restrict__ e,
                    const float* __restrict__ noise,
                    const float* __restrict__ gumbel,
                    const float* __restrict__ gate,
                    const float* __restrict__ ent,
                    const float* __restrict__ dstr,
                    __bf16* __restrict__ meas,
                    int Bn) {
    int b = blockIdx.x * blockDim.x + threadIdx.x;
    if (b >= Bn) return;

    const float* er = e + (size_t)b * 16;
    float amp[NQ], ph[NQ];
    float mx = -1e30f;
#pragma unroll
    for (int q = 0; q < NQ; ++q) { amp[q] = er[q]; mx = fmaxf(mx, amp[q]); }
    float s = 0.0f;
#pragma unroll
    for (int q = 0; q < NQ; ++q) { amp[q] = __expf(amp[q] - mx); s += amp[q]; }
    float inv = 1.0f / s;
#pragma unroll
    for (int q = 0; q < NQ; ++q) { amp[q] *= inv; ph[q] = er[NQ + q]; }

    float re[NQ], im[NQ];
#pragma unroll
    for (int q = 0; q < NQ; ++q) {
        re[q] = amp[q] * __cosf(ph[q]);
        im[q] = amp[q] * __sinf(ph[q]);
    }

    const float ds = dstr[0];
    for (int l = 0; l < NL; ++l) {
        for (int d = 0; d < NDEPTH; ++d) {
            const float* gp = gate + ((size_t)(l * NDEPTH + d) * NQ) * 3;
            const float* ep = ent + (size_t)(l * NDEPTH + d) * (NQ - 1);
            float nr[NQ], ni[NQ];
#pragma unroll
            for (int q = 0; q < NQ; ++q) {
                float ry = gp[q * 3 + 1], rz = gp[q * 3 + 2];
                float fac = __cosf(0.5f * ry);
                float c = __cosf(rz), sn = __sinf(rz);
                nr[q] = (re[q] * c - im[q] * sn) * fac;
                ni[q] = (re[q] * sn + im[q] * c) * fac;
            }
            re[0] = nr[0]; im[0] = ni[0];
#pragma unroll
            for (int q = 1; q < NQ; ++q) {
                float sg = 1.0f / (1.0f + __expf(-ep[q - 1]));
                re[q] = nr[q] + sg * nr[q - 1];
                im[q] = ni[q] + sg * ni[q - 1];
            }
        }
        const float* nz = noise + (((size_t)l * Bn + b) * NQ) * 2;
#pragma unroll
        for (int q = 0; q < NQ; ++q) {
            re[q] += ds * nz[q * 2];
            im[q] += ds * nz[q * 2 + 1];
        }
    }

    float p[NQ];
    mx = -1e30f;
#pragma unroll
    for (int q = 0; q < NQ; ++q) { p[q] = re[q] * re[q] + im[q] * im[q]; mx = fmaxf(mx, p[q]); }
    s = 0.0f;
#pragma unroll
    for (int q = 0; q < NQ; ++q) { p[q] = __expf(p[q] - mx); s += p[q]; }
    inv = 1.0f / s;

    float lg[NQ];
    mx = -1e30f;
#pragma unroll
    for (int q = 0; q < NQ; ++q) {
        float g = -__logf(-__logf(gumbel[(size_t)b * NQ + q] + 1e-10f) + 1e-10f);
        lg[q] = (__logf(p[q] * inv + 1e-10f) + g) * 2.0f;  // /TAU, TAU=0.5
        mx = fmaxf(mx, lg[q]);
    }
    s = 0.0f;
#pragma unroll
    for (int q = 0; q < NQ; ++q) { lg[q] = __expf(lg[q] - mx); s += lg[q]; }
    inv = 1.0f / s;

    __bf16* mp = meas + (size_t)b * 32;
#pragma unroll
    for (int q = 0; q < NQ; ++q) mp[q] = (__bf16)(lg[q] * inv);
#pragma unroll
    for (int q = NQ; q < 32; ++q) mp[q] = (__bf16)0.0f;
}

// ---------------------------------------------------------------------------
extern "C" void kernel_launch(void* const* d_in, const int* in_sizes, int n_in,
                              void* d_out, int out_size, void* d_ws, size_t ws_size,
                              hipStream_t stream) {
    (void)in_sizes; (void)n_in; (void)out_size; (void)ws_size;
    const float* x       = (const float*)d_in[0];
    const float* noise   = (const float*)d_in[2];
    const float* gumbel  = (const float*)d_in[3];
    const float* enc_w1  = (const float*)d_in[4];
    const float* enc_b1  = (const float*)d_in[5];
    const float* enc_w2  = (const float*)d_in[6];
    const float* enc_b2  = (const float*)d_in[7];
    const float* qe_w1   = (const float*)d_in[8];
    const float* qe_b1   = (const float*)d_in[9];
    const float* qe_w2   = (const float*)d_in[10];
    const float* qe_b2   = (const float*)d_in[11];
    const float* gate    = (const float*)d_in[12];
    const float* ent     = (const float*)d_in[13];
    const float* md_w1   = (const float*)d_in[14];
    const float* md_b1   = (const float*)d_in[15];
    const float* md_w2   = (const float*)d_in[16];
    const float* md_b2   = (const float*)d_in[17];
    const float* dec_w1  = (const float*)d_in[18];
    const float* dec_b1  = (const float*)d_in[19];
    const float* dec_w2  = (const float*)d_in[20];
    const float* dec_b2  = (const float*)d_in[21];
    const float* dstr    = (const float*)d_in[22];

    char* ws = (char*)d_ws;
    size_t off = 0;
    auto alloc = [&](size_t bytes) {
        size_t o = off;
        off = (off + bytes + 255) & ~(size_t)255;
        return (void*)(ws + o);
    };

    // transposed bf16 weights
    __bf16* wt_enc1 = (__bf16*)alloc((size_t)512 * 1024 * 2);
    __bf16* wt_enc2 = (__bf16*)alloc((size_t)256 * 512 * 2);
    __bf16* wt_qe1  = (__bf16*)alloc((size_t)64 * 256 * 2);
    __bf16* wt_qe2  = (__bf16*)alloc((size_t)16 * 64 * 2);
    __bf16* wt_md1  = (__bf16*)alloc((size_t)64 * 32 * 2);
    __bf16* wt_md2  = (__bf16*)alloc((size_t)256 * 64 * 2);
    __bf16* wt_dec1 = (__bf16*)alloc((size_t)512 * 256 * 2);
    __bf16* wt_dec2 = (__bf16*)alloc((size_t)1024 * 512 * 2);
    // activations (bf16 unless noted); reuse where lifetimes permit
    __bf16* a_h    = (__bf16*)alloc((size_t)BATCH * 512 * 2);  // h, later t3
    __bf16* a_feat = (__bf16*)alloc((size_t)BATCH * 256 * 2);  // feat, later qf
    __bf16* a_t1   = (__bf16*)alloc((size_t)BATCH * 64 * 2);   // t1, later t2
    float*  a_e    = (float*)alloc((size_t)BATCH * 16 * 4);
    __bf16* a_meas = (__bf16*)alloc((size_t)BATCH * 32 * 2);

    // ---- weight prep ----
    auto prep = [&](const float* W, __bf16* Wt, int K, int N, int Kpad) {
        int total = N * Kpad;
        prep_wt_kernel<<<(total + 255) / 256, 256, 0, stream>>>(W, Wt, K, N, Kpad);
    };
    prep(enc_w1, wt_enc1, 1024, 512, 1024);
    prep(enc_w2, wt_enc2, 512, 256, 512);
    prep(qe_w1,  wt_qe1,  256, 64, 256);
    prep(qe_w2,  wt_qe2,  64, 16, 64);
    prep(md_w1,  wt_md1,  8, 64, 32);     // K=8 zero-padded to 32
    prep(md_w2,  wt_md2,  64, 256, 64);
    prep(dec_w1, wt_dec1, 256, 512, 256);
    prep(dec_w2, wt_dec2, 512, 1024, 512);

    const dim3 blk(128);
    const int MB = BATCH / 64;  // 512 row-blocks

    // h = relu(x @ enc_w1 + b1)            [32768,1024]x[1024,512]
    gemm_wmma_kernel<8, true, true, false><<<dim3(MB, 512 / 128), blk, 0, stream>>>(
        x, wt_enc1, enc_b1, a_h, BATCH, 512, 1024);
    // feat = h @ enc_w2 + b2               [.,512]x[512,256]
    gemm_wmma_kernel<8, false, false, false><<<dim3(MB, 256 / 128), blk, 0, stream>>>(
        a_h, wt_enc2, enc_b2, a_feat, BATCH, 256, 512);
    // t1 = relu(feat @ qe_w1 + b1)         [.,256]x[256,64]
    gemm_wmma_kernel<4, true, false, false><<<dim3(MB, 1), blk, 0, stream>>>(
        a_feat, wt_qe1, qe_b1, a_t1, BATCH, 64, 256);
    // e = t1 @ qe_w2 + b2 (f32 out)        [.,64]x[64,16]
    gemm_wmma_kernel<1, false, false, true><<<dim3(MB, 1), blk, 0, stream>>>(
        a_t1, wt_qe2, qe_b2, a_e, BATCH, 16, 64);

    // quantum elementwise core -> meas (bf16, K-padded to 32)
    quantum_kernel<<<(BATCH + 255) / 256, 256, 0, stream>>>(
        a_e, noise, gumbel, gate, ent, dstr, a_meas, BATCH);

    // t2 = relu(meas @ md_w1 + b1)         [.,32(pad 8)]x[32,64]
    gemm_wmma_kernel<4, true, false, false><<<dim3(MB, 1), blk, 0, stream>>>(
        a_meas, wt_md1, md_b1, a_t1, BATCH, 64, 32);
    // qf = t2 @ md_w2 + b2                 [.,64]x[64,256]
    gemm_wmma_kernel<8, false, false, false><<<dim3(MB, 256 / 128), blk, 0, stream>>>(
        a_t1, wt_md2, md_b2, a_feat, BATCH, 256, 64);
    // t3 = relu(qf @ dec_w1 + b1)          [.,256]x[256,512]
    gemm_wmma_kernel<8, true, false, false><<<dim3(MB, 512 / 128), blk, 0, stream>>>(
        a_feat, wt_dec1, dec_b1, a_h, BATCH, 512, 256);
    // out = t3 @ dec_w2 + b2 (f32 out)     [.,512]x[512,1024]
    gemm_wmma_kernel<8, false, false, true><<<dim3(MB, 1024 / 128), blk, 0, stream>>>(
        a_h, wt_dec2, dec_b2, d_out, BATCH, 1024, 512);
}
